// RotaryMultiheadAttention_73581379715350
// MI455X (gfx1250) — compile-verified, matrix operand order checked by reading, not run on hardware
//
#include <hip/hip_runtime.h>
#include <hip/hip_bf16.h>

// ---------------------------------------------------------------------------
// RotaryMultiheadAttention forward for gfx1250 (MI455X), bf16 WMMA pipeline.
// GEMMs are LDS-staged with CDNA5 async global->LDS copies (ASYNCcnt).
// ---------------------------------------------------------------------------

typedef __bf16 bf16_t;
typedef __attribute__((ext_vector_type(16))) __bf16 v16bf;
typedef __attribute__((ext_vector_type(8)))  __bf16 v8bf;
typedef __attribute__((ext_vector_type(8)))  float  v8f;

#define EMBED   2048
#define NHEADS  16
#define HDIM    128
#define BATCH   2
#define SEQ     2048
#define MROWS   (BATCH * SEQ)     // 4096 token rows
#define NQKV    (3 * EMBED)       // 6144

// Staged-tile row stride in elements: 32 data + 8 pad = 80 bytes.
// Row r of a tile starts at bank (20*r) % 64 -> all 16 rows touched by one
// fragment read hit distinct bank groups (conflict-free b128 LDS reads).
#define ASTRIDE 40

// ---------------------------------------------------------------------------
// Fragment loader: row-major 16x32 bf16 tile -> per-lane v16bf in the CDNA5
// 16-bit A/B packing (lane = row%16; half = lane/16 selects K-chunks
// [half*8 .. half*8+7] and [16+half*8 .. 16+half*8+7]).
// ---------------------------------------------------------------------------
__device__ __forceinline__ v16bf load_frag(const bf16_t* __restrict__ p, int half) {
    const v8bf lo = *(const v8bf*)(p + half * 8);
    const v8bf hi = *(const v8bf*)(p + 16 + half * 8);
    return __builtin_shufflevector(lo, hi, 0, 1, 2, 3, 4, 5, 6, 7,
                                           8, 9, 10, 11, 12, 13, 14, 15);
}

__device__ __forceinline__ v8f wmma_bf16(v16bf a, v16bf b, v8f c) {
    return __builtin_amdgcn_wmma_f32_16x16x32_bf16(
        /*neg_a=*/false, a, /*neg_b=*/false, b,
        /*c_mod=*/(short)0, c, /*reuse_a=*/false, /*reuse_b=*/false);
}

// Async copy of one 16-byte chunk: global -> LDS, tracked by ASYNCcnt.
__device__ __forceinline__ void async_copy_b128(unsigned lds_off, const bf16_t* gaddr) {
    asm volatile("global_load_async_to_lds_b128 %0, %1, off"
                 :: "v"(lds_off), "v"(gaddr) : "memory");
}

// ---------------------------------------------------------------------------
// fp32 -> bf16 elementwise conversion (grid-stride)
// ---------------------------------------------------------------------------
__global__ void cvt_f32_to_bf16(const float* __restrict__ src,
                                bf16_t* __restrict__ dst, int n) {
    int i = blockIdx.x * blockDim.x + threadIdx.x;
    const int stride = gridDim.x * blockDim.x;
    for (; i < n; i += stride) dst[i] = (bf16_t)src[i];
}

// ---------------------------------------------------------------------------
// GEMM: C(MxN) = A(MxK) * Bm(NxK)^T + bias(N).  A, Bm row-major bf16.
// Workgroup = 256 threads (8 waves) -> 128x128 C tile; each wave does a
// 32x64 subtile = 2x4 WMMA tiles.  Double-buffered LDS staging via
// global_load_async_to_lds_b128 (4 async instrs / wave / k-step), overlapped
// with WMMA compute; s_wait_asynccnt gates buffer reuse.
// M,N multiples of 128; K multiple of 32.  Output bf16 (Cb) or f32 (Cf).
// ---------------------------------------------------------------------------
__global__ __launch_bounds__(256)
void wmma_gemm_bias(const bf16_t* __restrict__ A,
                    const bf16_t* __restrict__ Bm,
                    const float*  __restrict__ bias,
                    bf16_t* __restrict__ Cb,
                    float*  __restrict__ Cf,
                    int M, int N, int K) {
    __shared__ bf16_t smA[2][128 * ASTRIDE];   // 2 x 10 KB
    __shared__ bf16_t smB[2][128 * ASTRIDE];   // 2 x 10 KB

    const int tid  = threadIdx.x;
    const int lane = tid & 31;
    const int wave = tid >> 5;
    const int n16  = lane & 15;
    const int half = lane >> 4;
    const int wm   = wave & 3;          // 4 row-waves
    const int wn   = wave >> 2;         // 2 col-waves
    const int bm   = blockIdx.x * 128;
    const int bn   = blockIdx.y * 128;

    const bf16_t* gA = A  + (size_t)bm * K;
    const bf16_t* gB = Bm + (size_t)bn * K;

    // Stage one 128x32 A tile + 128x32 B tile into LDS buffer `buf`.
    // 512 16B-chunks per matrix; thread handles chunks {tid, tid+256}.
    auto prefetch = [&](int buf, int kk) {
        const unsigned baseA = (unsigned)(uintptr_t)&smA[buf][0];
        const unsigned baseB = (unsigned)(uintptr_t)&smB[buf][0];
#pragma unroll
        for (int i = 0; i < 2; ++i) {
            const int c   = tid + i * 256;
            const int row = c >> 2;          // 0..127
            const int col = (c & 3) * 8;     // element offset within 32-wide tile
            const unsigned loff = (unsigned)(row * ASTRIDE + col) * 2u;
            async_copy_b128(baseA + loff, gA + (size_t)row * K + kk + col);
            async_copy_b128(baseB + loff, gB + (size_t)row * K + kk + col);
        }
    };

    v8f acc[2][4];
#pragma unroll
    for (int tm = 0; tm < 2; ++tm)
#pragma unroll
        for (int tn = 0; tn < 4; ++tn) acc[tm][tn] = (v8f){};

    prefetch(0, 0);
    int buf = 0;
    for (int kk = 0; kk < K; kk += 32) {
        const bool hasNext = (kk + 32) < K;
        if (hasNext) {
            prefetch(buf ^ 1, kk + 32);
            // previous batch (4 async ops) done when count drops to the 4 just issued
            asm volatile("s_wait_asynccnt 0x4" ::: "memory");
        } else {
            asm volatile("s_wait_asynccnt 0x0" ::: "memory");
        }
        __syncthreads();   // all waves' copies for `buf` complete

        const bf16_t* sA = &smA[buf][0];
        const bf16_t* sB = &smB[buf][0];
        v16bf a[2];
#pragma unroll
        for (int tm = 0; tm < 2; ++tm)
            a[tm] = load_frag(sA + (wm * 32 + tm * 16 + n16) * ASTRIDE, half);
#pragma unroll
        for (int tn = 0; tn < 4; ++tn) {
            const v16bf bfr = load_frag(sB + (wn * 64 + tn * 16 + n16) * ASTRIDE, half);
#pragma unroll
            for (int tm = 0; tm < 2; ++tm)
                acc[tm][tn] = wmma_bf16(a[tm], bfr, acc[tm][tn]);
        }
        __syncthreads();   // everyone done reading `buf` before next overwrite
        buf ^= 1;
    }

#pragma unroll
    for (int tn = 0; tn < 4; ++tn) {
        const int gn = bn + wn * 64 + tn * 16 + n16;
        const float bv = bias[gn];
#pragma unroll
        for (int tm = 0; tm < 2; ++tm) {
            const int gm0 = bm + wm * 32 + tm * 16 + half * 8;
#pragma unroll
            for (int r = 0; r < 8; ++r) {
                const float v = acc[tm][tn][r] + bv;
                const size_t idx = (size_t)(gm0 + r) * N + gn;
                if (Cb) Cb[idx] = (bf16_t)v;
                else    Cf[idx] = v;
            }
        }
    }
}

// ---------------------------------------------------------------------------
// RoPE (interleaved pairs) + reorder:
//   qkv (B, L, 3, H, D) bf16  ->  Qr, Kr (B, H, L, D) bf16 roped,
//                                 Vt (B, H, D, L) bf16 (transposed for P*V)
// ---------------------------------------------------------------------------
__global__ __launch_bounds__(128)
void rope_reorder(const bf16_t* __restrict__ qkv,
                  bf16_t* __restrict__ Qr,
                  bf16_t* __restrict__ Kr,
                  bf16_t* __restrict__ Vt) {
    const int d = threadIdx.x;   // 0..127
    const int h = blockIdx.x;    // 0..15
    const int l = blockIdx.y;    // 0..2047
    const int b = blockIdx.z;    // 0..1

    const size_t rowbase = (((size_t)(b * SEQ + l)) * 3) * (NHEADS * HDIM)
                           + (size_t)h * HDIM + d;
    const float q = (float)qkv[rowbase];
    const float k = (float)qkv[rowbase + (size_t)(NHEADS * HDIM)];
    const float v = (float)qkv[rowbase + (size_t)(2 * NHEADS * HDIM)];

    const int pairbase = d & ~1;
    const float invf = __powf(10000.0f, -(float)pairbase / (float)HDIM);
    const float f = (float)l * invf;
    float s, c;
    __sincosf(f, &s, &c);

    const float qp  = __shfl_xor(q, 1, 32);   // partner within pair (d ^ 1)
    const float kp  = __shfl_xor(k, 1, 32);
    const float sgn = (d & 1) ? 1.0f : -1.0f; // even: x0*c - x1*s ; odd: x1*c + x0*s

    const size_t bh = (size_t)(b * NHEADS + h);
    Qr[(bh * SEQ + l) * HDIM + d] = (bf16_t)(q * c + sgn * qp * s);
    Kr[(bh * SEQ + l) * HDIM + d] = (bf16_t)(k * c + sgn * kp * s);
    Vt[(bh * HDIM + d) * SEQ + l] = (bf16_t)v;
}

// ---------------------------------------------------------------------------
// Causal flash attention.  One wave per 16 query rows (block = 4 waves = 64
// rows of one (b,h)).  Computes S^T = K * Q^T so each lane owns one query
// row: softmax row-reduction = in-lane over 8 regs + one xor-16 shuffle, and
// exp(P) lands exactly in the A-fragment slots for the P*V WMMA.
// ---------------------------------------------------------------------------
__global__ __launch_bounds__(128)
void flash_attn(const bf16_t* __restrict__ Qr,
                const bf16_t* __restrict__ Kr,
                const bf16_t* __restrict__ Vt,
                bf16_t* __restrict__ Obuf) {
    const int lane = threadIdx.x & 31;
    const int wave = threadIdx.x >> 5;
    const int n16  = lane & 15;
    const int half = lane >> 4;
    const int qb   = blockIdx.x * 64 + wave * 16;    // query tile base
    const int bh   = blockIdx.y;                     // b*H + h
    const int b    = bh >> 4;
    const int h    = bh & 15;

    const bf16_t* Qbh = Qr + (size_t)bh * SEQ * HDIM;
    const bf16_t* Kbh = Kr + (size_t)bh * SEQ * HDIM;
    const bf16_t* Vbh = Vt + (size_t)bh * HDIM * SEQ;

    // Q fragments: 16 rows x 128 d -> 4 K-fragments (used as WMMA "B")
    v16bf qf[4];
    {
        const bf16_t* qrow = Qbh + (size_t)(qb + n16) * HDIM;
#pragma unroll
        for (int kc = 0; kc < 4; ++kc) qf[kc] = load_frag(qrow + kc * 32, half);
    }

    v8f o[8];
#pragma unroll
    for (int f = 0; f < 8; ++f) o[f] = (v8f){};
    float m_run = -__builtin_inff();
    float l_run = 0.0f;
    const float scale = 0.08838834764831845f;   // 1/sqrt(128)
    const int qrow_idx = qb + n16;              // this lane's query row

    for (int s0 = 0; s0 <= qb + 15; s0 += 32) {
        // --- S^T tiles for keys [s0, s0+32) ---
        float xs[2][8];
#pragma unroll
        for (int sub = 0; sub < 2; ++sub) {
            int srow = s0 + sub * 16 + n16;
            if (srow > SEQ - 1) srow = SEQ - 1;
            const bf16_t* krow = Kbh + (size_t)srow * HDIM;
            v8f st = (v8f){};
#pragma unroll
            for (int kc = 0; kc < 4; ++kc)
                st = wmma_bf16(load_frag(krow + kc * 32, half), qf[kc], st);
#pragma unroll
            for (int r = 0; r < 8; ++r) {
                const int s = s0 + sub * 16 + half * 8 + r;
                float x = st[r] * scale;
                if (s > qrow_idx) x = -__builtin_inff();   // causal mask
                xs[sub][r] = x;
            }
        }

        // --- online softmax (per query row = per lane) ---
        float tmax = -__builtin_inff();
#pragma unroll
        for (int sub = 0; sub < 2; ++sub)
#pragma unroll
            for (int r = 0; r < 8; ++r) tmax = fmaxf(tmax, xs[sub][r]);
        tmax = fmaxf(tmax, __shfl_xor(tmax, 16, 32));

        const float newm = fmaxf(m_run, tmax);
        const float corr = __expf(m_run - newm);
        float tsum = 0.0f;
        v16bf pf;
#pragma unroll
        for (int sub = 0; sub < 2; ++sub)
#pragma unroll
            for (int r = 0; r < 8; ++r) {
                const float pv = __expf(xs[sub][r] - newm);
                tsum += pv;
                pf[sub * 8 + r] = (bf16_t)pv;   // exact A-fragment slot
            }
        tsum += __shfl_xor(tsum, 16, 32);
        l_run = l_run * corr + tsum;
        m_run = newm;

        // --- rescale O accumulators: per-row corr broadcast via shuffles ---
#pragma unroll
        for (int r = 0; r < 8; ++r) {
            const float cr = __shfl(corr, half * 8 + r, 32);
#pragma unroll
            for (int f = 0; f < 8; ++f) o[f][r] *= cr;
        }

        // --- O += P * V  (V transposed: rows = d, cols = s) ---
#pragma unroll
        for (int f = 0; f < 8; ++f) {
            const bf16_t* vp = Vbh + (size_t)(f * 16 + n16) * SEQ + s0;
            const v16bf vf = load_frag(vp, half);
            o[f] = wmma_bf16(pf, vf, o[f]);
        }
    }

    // --- epilogue: O /= l_run, write to (B, L, E) bf16 for out-projection ---
    const float linv = 1.0f / l_run;
#pragma unroll
    for (int r = 0; r < 8; ++r) {
        const float lr = __shfl(linv, half * 8 + r, 32);
        const int l = qb + half * 8 + r;
#pragma unroll
        for (int f = 0; f < 8; ++f) {
            const int dcol = f * 16 + n16;
            Obuf[((size_t)(b * SEQ + l)) * EMBED + h * HDIM + dcol] =
                (bf16_t)(o[f][r] * lr);
        }
    }
}

// ---------------------------------------------------------------------------
// Host-side launcher
// ---------------------------------------------------------------------------
extern "C" void kernel_launch(void* const* d_in, const int* in_sizes, int n_in,
                              void* d_out, int out_size, void* d_ws, size_t ws_size,
                              hipStream_t stream) {
    const float* x    = (const float*)d_in[0];  // (B, L, E)
    const float* wqkv = (const float*)d_in[1];  // (3E, E)
    const float* bqkv = (const float*)d_in[2];  // (3E,)
    const float* wout = (const float*)d_in[3];  // (E, E)
    const float* bout = (const float*)d_in[4];  // (E,)
    float* out = (float*)d_out;                 // (B, L, E) f32

    char* ws = (char*)d_ws;
    size_t off = 0;
    auto alloc = [&](size_t bytes) -> char* {
        char* p = ws + off;
        off = (off + bytes + 255) & ~(size_t)255;
        return p;
    };
    bf16_t* Xb    = (bf16_t*)alloc((size_t)MROWS * EMBED * 2);
    bf16_t* Wqkvb = (bf16_t*)alloc((size_t)NQKV * EMBED * 2);
    bf16_t* Woutb = (bf16_t*)alloc((size_t)EMBED * EMBED * 2);
    bf16_t* QKVb  = (bf16_t*)alloc((size_t)MROWS * NQKV * 2);
    bf16_t* Qrb   = (bf16_t*)alloc((size_t)BATCH * NHEADS * SEQ * HDIM * 2);
    bf16_t* Krb   = (bf16_t*)alloc((size_t)BATCH * NHEADS * SEQ * HDIM * 2);
    bf16_t* Vtb   = (bf16_t*)alloc((size_t)BATCH * NHEADS * HDIM * SEQ * 2);
    bf16_t* Ob    = (bf16_t*)alloc((size_t)MROWS * EMBED * 2);

    // 1) fp32 -> bf16 conversions
    cvt_f32_to_bf16<<<2048, 256, 0, stream>>>(x,    Xb,    MROWS * EMBED);
    cvt_f32_to_bf16<<<2048, 256, 0, stream>>>(wqkv, Wqkvb, NQKV * EMBED);
    cvt_f32_to_bf16<<<2048, 256, 0, stream>>>(wout, Woutb, EMBED * EMBED);

    // 2) fused QKV projection: (4096 x 6144) = Xb (4096 x 2048) * Wqkv^T
    wmma_gemm_bias<<<dim3(MROWS / 128, NQKV / 128), 256, 0, stream>>>(
        Xb, Wqkvb, bqkv, QKVb, nullptr, MROWS, NQKV, EMBED);

    // 3) RoPE + head reorder (+ V transpose)
    rope_reorder<<<dim3(NHEADS, SEQ, BATCH), HDIM, 0, stream>>>(QKVb, Qrb, Krb, Vtb);

    // 4) causal flash attention
    flash_attn<<<dim3(SEQ / 64, BATCH * NHEADS), 128, 0, stream>>>(Qrb, Krb, Vtb, Ob);

    // 5) output projection: d_out (4096 x 2048) = Ob * Wout^T + bout (f32 out)
    wmma_gemm_bias<<<dim3(MROWS / 128, EMBED / 128), 256, 0, stream>>>(
        Ob, Woutb, bout, nullptr, out, MROWS, EMBED, EMBED);

    (void)in_sizes; (void)n_in; (void)out_size; (void)ws_size;
}
